// ADLS_13022340842024
// MI455X (gfx1250) — matrix operand order repaired and checked by reading, not compiled, not run on hardware
//
#include <hip/hip_runtime.h>
#include <hip/hip_bf16.h>

// ---------------------------------------------------------------------------
// CDNA5 (gfx1250): bf16 WMMA backbone + routed LoRA, async-LDS double buffering
// ---------------------------------------------------------------------------

typedef __attribute__((ext_vector_type(16))) __bf16 v16bf;
typedef __attribute__((ext_vector_type(8)))  __bf16 v8bf;
typedef __attribute__((ext_vector_type(4)))  __bf16 v4bf;
typedef __attribute__((ext_vector_type(8)))  float  v8f;

#define LORA_SCALING 0.25f
#define EPSF    1e-8f
#define EPS_LNF 1e-5f

#define WMMA_BF16(a, b, c) \
  __builtin_amdgcn_wmma_f32_16x16x32_bf16(false, (a), false, (b), (short)0, (c), false, false)

// Async global->LDS copy path (GLOBAL_LOAD_ASYNC_TO_LDS_*, ASYNCcnt).
// Probe-learned signature: (intN-vec AS1*, intN-vec AS3*, imm offset, imm cpol).
#if defined(__AMDGCN__) && __has_builtin(__builtin_amdgcn_global_load_async_to_lds_b128)
#define HAVE_ASYNC_LDS 1
typedef int v4i_ __attribute__((vector_size(16)));
#define ASYNC_B128(gp, lp)                                          \
  __builtin_amdgcn_global_load_async_to_lds_b128(                   \
      (__attribute__((address_space(1))) v4i_*)(gp),                \
      (__attribute__((address_space(3))) v4i_*)(lp), 0, 0)
#if __has_builtin(__builtin_amdgcn_global_load_async_to_lds_b64)
#define HAVE_ASYNC_LDS_B64 1
typedef int v2i_ __attribute__((vector_size(8)));
#define ASYNC_B64(gp, lp)                                           \
  __builtin_amdgcn_global_load_async_to_lds_b64(                    \
      (__attribute__((address_space(1))) v2i_*)(gp),                \
      (__attribute__((address_space(3))) v2i_*)(lp), 0, 0)
#endif
#if __has_builtin(__builtin_amdgcn_s_wait_asynccnt)
#define ASYNC_WAIT_(n) __builtin_amdgcn_s_wait_asynccnt(n)
#else
#define ASYNC_WAIT_(n) asm volatile("s_wait_asynccnt " #n ::: "memory")
#endif
#define ASYNC_WAIT(n) ASYNC_WAIT_(n)   // two-level: expand macro args first
#endif

// per-K-step async groups per wave
#ifdef HAVE_ASYNC_LDS_B64
#define LORA_GROUP 3
#else
#define LORA_GROUP 2
#endif
#define GEMM_GROUP 3

// A-operand fragment (16x32): elems[0..7]=K(half*8..+7), elems[8..15]=K(16+half*8..+7)
static __device__ __forceinline__ v16bf mk_a(const __bf16* rowPtr, int half) {
  v8bf lo = *(const v8bf*)(rowPtr + half * 8);
  v8bf hi = *(const v8bf*)(rowPtr + 16 + half * 8);
  return __builtin_shufflevector(lo, hi, 0,1,2,3,4,5,6,7,8,9,10,11,12,13,14,15);
}

// B-operand fragment (32x16, lane = column n): elems[0..7]=K(half*8..+7), elems[8..15]=K(16+half*8..+7)
static __device__ __forceinline__ v16bf mk_b(const __bf16* rowPtr, int half) {
  v8bf lo = *(const v8bf*)(rowPtr + half * 8);
  v8bf hi = *(const v8bf*)(rowPtr + 16 + half * 8);
  return __builtin_shufflevector(lo, hi, 0,1,2,3,4,5,6,7,8,9,10,11,12,13,14,15);
}

// ---------------------------------------------------------------------------
// fp32 -> bf16 elementwise convert
// ---------------------------------------------------------------------------
__global__ void cvt_bf16_kernel(const float* __restrict__ in, __bf16* __restrict__ out, long n) {
  long t = (long)blockIdx.x * 256 + threadIdx.x;
  if (t < n) out[t] = (__bf16)in[t];
}

// Bm [E=8][outDim][R=4] fp32 -> BmFlat [outDim][32] bf16, col c = e*4+r
__global__ void bmflat_kernel(const float* __restrict__ Bm, __bf16* __restrict__ out, int outDim) {
  long t = (long)blockIdx.x * 256 + threadIdx.x;
  long total = (long)outDim * 32;
  if (t >= total) return;
  int c = (int)(t & 31);
  long n = t >> 5;
  int e = c >> 2, r = c & 3;
  out[t] = (__bf16)Bm[((long)e * outDim + n) * 4 + r];
}

// ---------------------------------------------------------------------------
// Embedding gather: x_bf16[b, f*32+j] = bf16(emb[field_idx[b,f]][j])
// ---------------------------------------------------------------------------
__global__ void embed_kernel(const int* __restrict__ field_idx,
                             const float* __restrict__ emb,
                             __bf16* __restrict__ xb) {
  long t = (long)blockIdx.x * 256 + threadIdx.x;  // B*F*4 chunks of 8 floats
  int  j8 = (int)(t & 3) * 8;
  long bf = t >> 2;
  long fid = field_idx[bf];
  const float* src = emb + fid * 32 + j8;
  __bf16* dst = xb + bf * 32 + j8;
#pragma unroll
  for (int j = 0; j < 8; ++j) dst[j] = (__bf16)src[j];
}

// ---------------------------------------------------------------------------
// Routing: ri depends on b only through domain_id -> 24-row problem.
// Produces zeta[D*3], alpha[D*3*8].
// ---------------------------------------------------------------------------
__global__ void routing_kernel(const float* __restrict__ dom_emb,
                               const float* __restrict__ layer_pos,
                               const float* __restrict__ gate_logits,
                               const float* __restrict__ Wi1, const float* __restrict__ bi1,
                               const float* __restrict__ gi,  const float* __restrict__ bti,
                               const float* __restrict__ Wi2, const float* __restrict__ bi2,
                               const float* __restrict__ Wr1, const float* __restrict__ br1,
                               const float* __restrict__ gr,  const float* __restrict__ btr,
                               const float* __restrict__ Wr2, const float* __restrict__ br2,
                               float* __restrict__ zeta, float* __restrict__ alpha) {
  int d = threadIdx.x;
  if (d >= 8) return;
  float g = logf(1.0f + expf(gate_logits[d * 8 + d]));
  float wdd = g / fmaxf(g, EPSF);
  float hp[64];
  for (int j = 0; j < 64; ++j) hp[j] = wdd * dom_emb[d * 64 + j];

  float scores[3];
  float pr[3][8];
  for (int l = 0; l < 3; ++l) {
    float ri[96];
    for (int j = 0; j < 64; ++j) ri[j] = hp[j];
    for (int j = 0; j < 32; ++j) ri[64 + j] = layer_pos[l * 32 + j];

    float z[64];
    for (int o = 0; o < 64; ++o) {
      float a = bi1[o];
      for (int k = 0; k < 96; ++k) a += ri[k] * Wi1[o * 96 + k];
      z[o] = a;
    }
    float m = 0.f; for (int o = 0; o < 64; ++o) m += z[o]; m *= (1.f / 64.f);
    float v = 0.f; for (int o = 0; o < 64; ++o) { float t = z[o] - m; v += t * t; } v *= (1.f / 64.f);
    float rs = rsqrtf(v + EPS_LNF);
    float sc = bi2[0];
    for (int o = 0; o < 64; ++o) {
      float h = fmaxf(gi[o] * (z[o] - m) * rs + bti[o], 0.f);
      sc += h * Wi2[o];
    }
    scores[l] = sc;

    float zr[64];
    for (int o = 0; o < 64; ++o) {
      float a = br1[o];
      for (int k = 0; k < 96; ++k) a += ri[k] * Wr1[o * 96 + k];
      zr[o] = a;
    }
    m = 0.f; for (int o = 0; o < 64; ++o) m += zr[o]; m *= (1.f / 64.f);
    v = 0.f; for (int o = 0; o < 64; ++o) { float t = zr[o] - m; v += t * t; } v *= (1.f / 64.f);
    rs = rsqrtf(v + EPS_LNF);
    float hr[64];
    for (int o = 0; o < 64; ++o) hr[o] = fmaxf(gr[o] * (zr[o] - m) * rs + btr[o], 0.f);
    float e[8];
    float emax = -1e30f;
    for (int q = 0; q < 8; ++q) {
      float a = br2[q];
      for (int o = 0; o < 64; ++o) a += hr[o] * Wr2[q * 64 + o];
      e[q] = a; emax = fmaxf(emax, a);
    }
    float es = 0.f;
    for (int q = 0; q < 8; ++q) { e[q] = expf(e[q] - emax); es += e[q]; }
    for (int q = 0; q < 8; ++q) e[q] /= es;
    int i1 = 0; for (int q = 1; q < 8; ++q) if (e[q] > e[i1]) i1 = q;
    int i2 = -1;
    for (int q = 0; q < 8; ++q) { if (q == i1) continue; if (i2 < 0 || e[q] > e[i2]) i2 = q; }
    float s = fmaxf(e[i1] + e[i2], EPSF);
    for (int q = 0; q < 8; ++q) pr[l][q] = (q == i1 || q == i2) ? e[q] / s : 0.f;
  }

  float smax = fmaxf(scores[0], fmaxf(scores[1], scores[2]));
  float p[3], ps = 0.f;
  for (int l = 0; l < 3; ++l) { p[l] = expf(scores[l] - smax); ps += p[l]; }
  for (int l = 0; l < 3; ++l) p[l] /= ps;
  int dm = 0;
  for (int l = 1; l < 3; ++l) if (p[l] <= p[dm]) dm = l;
  float zs = 0.f;
  for (int l = 0; l < 3; ++l) if (l != dm) zs += p[l];
  zs = fmaxf(zs, EPSF);
  for (int l = 0; l < 3; ++l) zeta[d * 3 + l] = (l != dm) ? p[l] / zs : 0.f;
  for (int l = 0; l < 3; ++l)
    for (int q = 0; q < 8; ++q) alpha[(d * 3 + l) * 8 + q] = pr[l][q];
}

// ---------------------------------------------------------------------------
// LoRA "t" pass: U[b, e*4+r] = SCALING*zeta[d,l]*alpha[d,l,e] * (H @ Aall.T)
// M x K x 32 WMMA GEMM, double-buffered async LDS staging (branch-free loop).
// ---------------------------------------------------------------------------
__global__ __launch_bounds__(256)
void lora_t_kernel(const __bf16* __restrict__ H, const __bf16* __restrict__ Aall,
                   const int* __restrict__ domain_id, const float* __restrict__ zeta,
                   const float* __restrict__ alpha, int layer, int K,
                   __bf16* __restrict__ U) {
  __shared__ __align__(16) __bf16 lA[2][128 * 40];
  __shared__ __align__(16) __bf16 lB[2][32 * 40];
  __shared__ __align__(16) __bf16 lU[128 * 40];
  const int tid  = threadIdx.x;
  const int wid  = tid >> 5;
  const int lane = tid & 31;
  const int half = lane >> 4;
  const int ln16 = lane & 15;
  const long rowBase = (long)blockIdx.x * 128;

  const int aR = tid >> 2, aC = (tid & 3) << 3;   // A staging: 2 x b128 / thread
  const int bR = tid >> 3, bC = (tid & 7) << 2;   // B staging: 1 x b64  / thread

  auto stage = [&](int buf, int k0) {
#ifdef HAVE_ASYNC_LDS
    ASYNC_B128(&H[(rowBase + aR) * (long)K + k0 + aC],      &lA[buf][aR * 40 + aC]);
    ASYNC_B128(&H[(rowBase + 64 + aR) * (long)K + k0 + aC], &lA[buf][(64 + aR) * 40 + aC]);
#ifdef HAVE_ASYNC_LDS_B64
    ASYNC_B64(&Aall[(long)bR * K + k0 + bC], &lB[buf][bR * 40 + bC]);
#else
    *(v4bf*)&lB[buf][bR * 40 + bC] = *(const v4bf*)&Aall[(long)bR * K + k0 + bC];
#endif
#else
    *(v8bf*)&lA[buf][aR * 40 + aC]        = *(const v8bf*)&H[(rowBase + aR) * (long)K + k0 + aC];
    *(v8bf*)&lA[buf][(64 + aR) * 40 + aC] = *(const v8bf*)&H[(rowBase + 64 + aR) * (long)K + k0 + aC];
    *(v4bf*)&lB[buf][bR * 40 + bC]        = *(const v4bf*)&Aall[(long)bR * K + k0 + bC];
#endif
  };

  v8f c0 = {}; v8f c1 = {};
  auto compute = [&](int cur) {
    v16bf a  = mk_a(&lA[cur][(wid * 16 + ln16) * 40], half);
    v16bf b0 = mk_b(&lB[cur][ln16 * 40], half);
    v16bf b1 = mk_b(&lB[cur][(16 + ln16) * 40], half);
    c0 = WMMA_BF16(a, b0, c0);
    c1 = WMMA_BF16(a, b1, c1);
  };

  const int nsteps = K >> 5;
  stage(0, 0);
  for (int s = 0; s < nsteps - 1; ++s) {           // branch-free steady state
    int cur = s & 1;
    stage(cur ^ 1, (s + 1) << 5);
#ifdef HAVE_ASYNC_LDS
    ASYNC_WAIT(LORA_GROUP);
#endif
    __syncthreads();
    compute(cur);
    __syncthreads();
  }
#ifdef HAVE_ASYNC_LDS
  ASYNC_WAIT(0);
#endif
  __syncthreads();
  compute((nsteps - 1) & 1);

  // scale by SCALING*zeta*alpha, bounce through LDS, coalesced b128 store
  const int rloc = wid * 16 + half * 8;
  const int e0 = ln16 >> 2;
  const int e1 = (16 + ln16) >> 2;
#pragma unroll
  for (int i = 0; i < 8; ++i) {
    int d = domain_id[rowBase + rloc + i];
    float zl = LORA_SCALING * zeta[d * 3 + layer];
    const float* al = alpha + (d * 3 + layer) * 8;
    lU[(rloc + i) * 40 + ln16]      = (__bf16)(c0[i] * zl * al[e0]);
    lU[(rloc + i) * 40 + 16 + ln16] = (__bf16)(c1[i] * zl * al[e1]);
  }
  __syncthreads();
#pragma unroll
  for (int it = 0; it < 2; ++it) {
    int idx = tid + it * 256;            // 512 chunks of 8
    int r = idx >> 2, c8 = (idx & 3) << 3;
    *(v8bf*)&U[(rowBase + r) * 32 + c8] = *(const v8bf*)&lU[r * 40 + c8];
  }
}

// ---------------------------------------------------------------------------
// Backbone layer: Out = relu(H @ W.T + bias + U @ BmFlat.T), bf16 out.
// Block tile 128x64, 8 waves (4x2), wave tile 32x32 (2x2 WMMA).
// Double-buffered async LDS staging; LoRA = one extra K=32 WMMA per tile.
// ---------------------------------------------------------------------------
__global__ __launch_bounds__(256)
void layer_gemm_kernel(const __bf16* __restrict__ H, const __bf16* __restrict__ Wb,
                       const float* __restrict__ bias, const __bf16* __restrict__ U,
                       const __bf16* __restrict__ BmF, __bf16* __restrict__ Out,
                       int K, int N) {
  __shared__ __align__(16) __bf16 lA[2][128 * 40];
  __shared__ __align__(16) __bf16 lB[2][64 * 40];
  __shared__ __align__(16) __bf16 lO[128 * 72];
  const int tid  = threadIdx.x;
  const int wid  = tid >> 5;
  const int lane = tid & 31;
  const int half = lane >> 4;
  const int ln16 = lane & 15;
  const int wm = wid >> 1, wn = wid & 1;
  const long rowBase = (long)blockIdx.x * 128;
  const long colBase = (long)blockIdx.y * 64;

  const int aR = tid >> 2, aC = (tid & 3) << 3;   // 16B chunk coords

  auto stage = [&](int buf, int k0) {
#ifdef HAVE_ASYNC_LDS
    ASYNC_B128(&H[(rowBase + aR) * (long)K + k0 + aC],      &lA[buf][aR * 40 + aC]);
    ASYNC_B128(&H[(rowBase + 64 + aR) * (long)K + k0 + aC], &lA[buf][(64 + aR) * 40 + aC]);
    ASYNC_B128(&Wb[(colBase + aR) * (long)K + k0 + aC],     &lB[buf][aR * 40 + aC]);
#else
    *(v8bf*)&lA[buf][aR * 40 + aC]        = *(const v8bf*)&H[(rowBase + aR) * (long)K + k0 + aC];
    *(v8bf*)&lA[buf][(64 + aR) * 40 + aC] = *(const v8bf*)&H[(rowBase + 64 + aR) * (long)K + k0 + aC];
    *(v8bf*)&lB[buf][aR * 40 + aC]        = *(const v8bf*)&Wb[(colBase + aR) * (long)K + k0 + aC];
#endif
  };

  v8f c00 = {}, c01 = {}, c10 = {}, c11 = {};
  auto compute = [&](int cur) {
    v16bf a0 = mk_a(&lA[cur][(wm * 32 + ln16) * 40], half);
    v16bf a1 = mk_a(&lA[cur][(wm * 32 + 16 + ln16) * 40], half);
    v16bf b0 = mk_b(&lB[cur][(wn * 32 + ln16) * 40], half);
    v16bf b1 = mk_b(&lB[cur][(wn * 32 + 16 + ln16) * 40], half);
    c00 = WMMA_BF16(a0, b0, c00);
    c01 = WMMA_BF16(a0, b1, c01);
    c10 = WMMA_BF16(a1, b0, c10);
    c11 = WMMA_BF16(a1, b1, c11);
  };

  const int nsteps = K >> 5;
  stage(0, 0);
  for (int s = 0; s < nsteps - 1; ++s) {           // branch-free steady state
    int cur = s & 1;
    stage(cur ^ 1, (s + 1) << 5);
#ifdef HAVE_ASYNC_LDS
    ASYNC_WAIT(GEMM_GROUP);
#endif
    __syncthreads();
    compute(cur);
    __syncthreads();
  }
#ifdef HAVE_ASYNC_LDS
  ASYNC_WAIT(0);
#endif
  __syncthreads();
  compute((nsteps - 1) & 1);

  // fused routed-LoRA projection: one extra K=32 WMMA per tile (direct global)
  {
    long r0 = rowBase + wm * 32;
    long n0 = colBase + wn * 32;
    v16bf ua0 = mk_a(&U[(r0 + ln16) * 32], half);
    v16bf ua1 = mk_a(&U[(r0 + 16 + ln16) * 32], half);
    v16bf ub0 = mk_b(&BmF[(n0 + ln16) * 32], half);
    v16bf ub1 = mk_b(&BmF[(n0 + 16 + ln16) * 32], half);
    c00 = WMMA_BF16(ua0, ub0, c00);
    c01 = WMMA_BF16(ua0, ub1, c01);
    c10 = WMMA_BF16(ua1, ub0, c10);
    c11 = WMMA_BF16(ua1, ub1, c11);
  }

  // bias + relu into LDS, then coalesced b128 row stores
  const long n0g = colBase + wn * 32 + ln16;
  const float bias0 = bias[n0g];
  const float bias1 = bias[n0g + 16];
  const int rloc = wm * 32 + half * 8;
  const int cloc = wn * 32 + ln16;
#pragma unroll
  for (int i = 0; i < 8; ++i) {
    lO[(rloc + i) * 72 + cloc]           = (__bf16)fmaxf(c00[i] + bias0, 0.f);
    lO[(rloc + i) * 72 + cloc + 16]      = (__bf16)fmaxf(c01[i] + bias1, 0.f);
    lO[(rloc + 16 + i) * 72 + cloc]      = (__bf16)fmaxf(c10[i] + bias0, 0.f);
    lO[(rloc + 16 + i) * 72 + cloc + 16] = (__bf16)fmaxf(c11[i] + bias1, 0.f);
  }
  __syncthreads();
#pragma unroll
  for (int it = 0; it < 4; ++it) {
    int idx = tid + it * 256;            // 1024 chunks of 8
    int r = idx >> 3, c8 = (idx & 7) << 3;
    *(v8bf*)&Out[(rowBase + r) * (long)N + colBase + c8] = *(const v8bf*)&lO[r * 72 + c8];
  }
}

// ---------------------------------------------------------------------------
// Per-domain tower: one wave per sample, coalesced reads, shfl_xor reduction
// ---------------------------------------------------------------------------
__global__ __launch_bounds__(256)
void tower_kernel(const __bf16* __restrict__ H2, const int* __restrict__ dom,
                  const float* __restrict__ Wt1, const float* __restrict__ bt1,
                  const float* __restrict__ Wt2, const float* __restrict__ bt2,
                  float* __restrict__ out) {
  const int lane = threadIdx.x & 31;
  const int wid  = threadIdx.x >> 5;
  const long b = (long)blockIdx.x * 8 + wid;
  const int d = dom[b];
  const __bf16* h = H2 + b * 512 + lane * 16;
  float hv[16];
  {
    v8bf p0 = *(const v8bf*)(h);
    v8bf p1 = *(const v8bf*)(h + 8);
#pragma unroll
    for (int j = 0; j < 8; ++j) { hv[j] = (float)p0[j]; hv[8 + j] = (float)p1[j]; }
  }
  const float* w = Wt1 + (long)d * 8 * 512 + lane * 16;
  float acc[8];
#pragma unroll
  for (int o = 0; o < 8; ++o) {
    float a = 0.f;
    const float* wo = w + o * 512;
#pragma unroll
    for (int j = 0; j < 16; ++j) a += hv[j] * wo[j];
    acc[o] = a;
  }
#pragma unroll
  for (int o = 0; o < 8; ++o)
#pragma unroll
    for (int off = 16; off > 0; off >>= 1)
      acc[o] += __shfl_xor(acc[o], off, 32);
  if (lane == 0) {
    float logit = bt2[d];
#pragma unroll
    for (int o = 0; o < 8; ++o)
      logit += Wt2[d * 8 + o] * fmaxf(acc[o] + bt1[d * 8 + o], 0.f);
    out[b] = logit;
  }
}

// ---------------------------------------------------------------------------
// launch
// ---------------------------------------------------------------------------
extern "C" void kernel_launch(void* const* d_in, const int* in_sizes, int n_in,
                              void* d_out, int out_size, void* d_ws, size_t ws_size,
                              hipStream_t stream) {
  (void)in_sizes; (void)n_in; (void)out_size; (void)ws_size;
  const int B = 16384;
  const int IN = 1024, N0 = 2048, N1 = 1024, N2 = 512;

  const int*   field_idx  = (const int*)  d_in[0];
  const int*   domain_id  = (const int*)  d_in[1];
  const float* emb_table  = (const float*)d_in[2];
  const float* W0 = (const float*)d_in[3];  const float* b0 = (const float*)d_in[4];
  const float* W1 = (const float*)d_in[5];  const float* b1 = (const float*)d_in[6];
  const float* W2 = (const float*)d_in[7];  const float* b2 = (const float*)d_in[8];
  const float* A0 = (const float*)d_in[9];  const float* Bm0 = (const float*)d_in[10];
  const float* A1 = (const float*)d_in[11]; const float* Bm1 = (const float*)d_in[12];
  const float* A2 = (const float*)d_in[13]; const float* Bm2 = (const float*)d_in[14];
  const float* dom_emb    = (const float*)d_in[15];
  const float* layer_pos  = (const float*)d_in[16];
  const float* gate_logit = (const float*)d_in[17];
  const float* Wi1 = (const float*)d_in[18]; const float* bi1 = (const float*)d_in[19];
  const float* gi  = (const float*)d_in[20]; const float* bti = (const float*)d_in[21];
  const float* Wi2 = (const float*)d_in[22]; const float* bi2 = (const float*)d_in[23];
  const float* Wr1 = (const float*)d_in[24]; const float* br1 = (const float*)d_in[25];
  const float* gr  = (const float*)d_in[26]; const float* btr = (const float*)d_in[27];
  const float* Wr2 = (const float*)d_in[28]; const float* br2 = (const float*)d_in[29];
  const float* Wt1 = (const float*)d_in[30]; const float* bt1 = (const float*)d_in[31];
  const float* Wt2 = (const float*)d_in[32]; const float* bt2 = (const float*)d_in[33];

  char* p = (char*)d_ws;
  auto carve = [&](size_t bytes) -> void* {
    void* r = (void*)p;
    p += (bytes + 255) & ~(size_t)255;
    return r;
  };
  __bf16* xb   = (__bf16*)carve((size_t)B * IN * 2);
  __bf16* h0b  = (__bf16*)carve((size_t)B * N0 * 2);
  __bf16* h1b  = (__bf16*)carve((size_t)B * N1 * 2);
  __bf16* h2b  = (__bf16*)carve((size_t)B * N2 * 2);
  __bf16* ub   = (__bf16*)carve((size_t)B * 32 * 2);
  __bf16* W0b  = (__bf16*)carve((size_t)N0 * IN * 2);
  __bf16* W1b  = (__bf16*)carve((size_t)N1 * N0 * 2);
  __bf16* W2b  = (__bf16*)carve((size_t)N2 * N1 * 2);
  __bf16* A0b  = (__bf16*)carve((size_t)32 * IN * 2);
  __bf16* A1b  = (__bf16*)carve((size_t)32 * N0 * 2);
  __bf16* A2b  = (__bf16*)carve((size_t)32 * N1 * 2);
  __bf16* Bm0F = (__bf16*)carve((size_t)N0 * 32 * 2);
  __bf16* Bm1F = (__bf16*)carve((size_t)N1 * 32 * 2);
  __bf16* Bm2F = (__bf16*)carve((size_t)N2 * 32 * 2);
  float*  zeta  = (float*)carve(24 * 4);
  float*  alpha = (float*)carve(192 * 4);

  auto cvt = [&](const float* in, __bf16* out, long n) {
    cvt_bf16_kernel<<<(unsigned)((n + 255) / 256), 256, 0, stream>>>(in, out, n);
  };
  cvt(W0, W0b, (long)N0 * IN);
  cvt(W1, W1b, (long)N1 * N0);
  cvt(W2, W2b, (long)N2 * N1);
  cvt(A0, A0b, 32L * IN);
  cvt(A1, A1b, 32L * N0);
  cvt(A2, A2b, 32L * N1);
  bmflat_kernel<<<(N0 * 32 + 255) / 256, 256, 0, stream>>>(Bm0, Bm0F, N0);
  bmflat_kernel<<<(N1 * 32 + 255) / 256, 256, 0, stream>>>(Bm1, Bm1F, N1);
  bmflat_kernel<<<(N2 * 32 + 255) / 256, 256, 0, stream>>>(Bm2, Bm2F, N2);

  routing_kernel<<<1, 8, 0, stream>>>(dom_emb, layer_pos, gate_logit,
                                      Wi1, bi1, gi, bti, Wi2, bi2,
                                      Wr1, br1, gr, btr, Wr2, br2, zeta, alpha);

  embed_kernel<<<(B * 32 * 4) / 256, 256, 0, stream>>>(field_idx, emb_table, xb);

  // layer 0: 1024 -> 2048
  lora_t_kernel<<<B / 128, 256, 0, stream>>>(xb, A0b, domain_id, zeta, alpha, 0, IN, ub);
  layer_gemm_kernel<<<dim3(B / 128, N0 / 64), 256, 0, stream>>>(xb, W0b, b0, ub, Bm0F, h0b, IN, N0);
  // layer 1: 2048 -> 1024
  lora_t_kernel<<<B / 128, 256, 0, stream>>>(h0b, A1b, domain_id, zeta, alpha, 1, N0, ub);
  layer_gemm_kernel<<<dim3(B / 128, N1 / 64), 256, 0, stream>>>(h0b, W1b, b1, ub, Bm1F, h1b, N0, N1);
  // layer 2: 1024 -> 512
  lora_t_kernel<<<B / 128, 256, 0, stream>>>(h1b, A2b, domain_id, zeta, alpha, 2, N1, ub);
  layer_gemm_kernel<<<dim3(B / 128, N2 / 64), 256, 0, stream>>>(h1b, W2b, b2, ub, Bm2F, h2b, N1, N2);

  tower_kernel<<<B / 8, 256, 0, stream>>>(h2b, domain_id, Wt1, bt1, Wt2, bt2, (float*)d_out);
}
